// MixLoraProjLayer_10015863734800
// MI455X (gfx1250) — matrix operand order repaired and checked by reading, not compiled, not run on hardware
//
#include <hip/hip_runtime.h>
#include <hip/hip_bf16.h>

typedef __attribute__((ext_vector_type(16))) __bf16 v16bf;
typedef __attribute__((ext_vector_type(4)))  __bf16 v4bf;
typedef __attribute__((ext_vector_type(8)))  float  v8f;
typedef __attribute__((ext_vector_type(4)))  float  f4;

#define DK    4096     // D_IN
#define EN    8        // experts
#define RN    16       // LoRA rank (N per expert)
#define MTOT  16384    // B*S tokens
#define WROW  40       // padded LDS row stride (bf16 elems): 80B rows -> conflict-free b128 reads

__device__ __forceinline__ __bf16 f2bf(float f) { return (__bf16)f; }

// ---------------------------------------------------------------------------
// Kernel 1: convert W (E*R*DK fp32) -> bf16 in workspace, [e][r][k] layout.
// ---------------------------------------------------------------------------
__global__ __launch_bounds__(256) void w_cvt_bf16(const float* __restrict__ W,
                                                  __bf16* __restrict__ Wb) {
    int i = blockIdx.x * blockDim.x + threadIdx.x;   // 0 .. 131071
    f4 v = ((const f4*)W)[i];
    v4bf o;
    o[0] = f2bf(v.x); o[1] = f2bf(v.y); o[2] = f2bf(v.z); o[3] = f2bf(v.w);
    ((v4bf*)Wb)[i] = o;
}

// ---------------------------------------------------------------------------
// Kernel 2: grouped LoRA GEMM.
//   - wave32 handles M=32 tokens (two 16x16x32 A fragments), all 8 experts
//   - W K-slice (8 KB) staged into LDS once per block, double buffered
//   - B fragments pipelined through TWO live register buffers (bA/bB) so
//     ds_load latency is covered by WMMA issues instead of s_wait_dscnt 0
//   - x read exactly once from HBM
// ---------------------------------------------------------------------------
__global__ __launch_bounds__(256) void lora_wmma_bf16(const float*  __restrict__ X,
                                                      const __bf16* __restrict__ Wb,
                                                      float*        __restrict__ Out) {
    __shared__ __bf16 smem[2][EN * RN * WROW];   // 2 x 10240 B

    const int t      = threadIdx.x;
    const int lane   = t & 31;
    const int wave   = t >> 5;
    const int waveId = blockIdx.x * 8 + wave;    // 0..511
    const int m0     = waveId * 32;

    const int half = lane >> 4;                  // 0: lanes 0-15, 1: lanes 16-31
    const int lmod = lane & 15;

    // ---- staging role: thread t copies 16 bf16 of row (se, sr), k-half sh ----
    const int se = t >> 5;                       // expert 0..7
    const int sr = (t >> 1) & 15;                // row r 0..15
    const int sh = t & 1;                        // k-half 0..1
    const __bf16* wsrc = Wb + ((size_t)(se * RN + sr)) * DK + sh * 16;
    const int stageOff = (se * RN + sr) * WROW + sh * 16;

    // ---- compute roles ----
    const float* xrow0 = X + (size_t)(m0 + lmod) * DK;        // tile 0 row
    const float* xrow1 = X + (size_t)(m0 + 16 + lmod) * DK;   // tile 1 row
    const int akoff = half * 16;   // A: this lane covers K = k0+akoff..+15
    const int boff  = lmod * WROW + half * 16;   // B fragment offset within expert slab

    v8f acc0[EN] = {};
    v8f acc1[EN] = {};

    // prime LDS stage 0 with W slice for k0 = 0
    { v16bf w = *(const v16bf*)(wsrc); *(v16bf*)&smem[0][stageOff] = w; }
    __syncthreads();

#pragma unroll 1
    for (int k0 = 0; k0 < DK; k0 += 32) {
        const int  p       = (k0 >> 5) & 1;
        const bool hasNext = (k0 + 32) < DK;

        // issue next W slice's global load first (consumed by ds_store at loop end)
        v16bf wnext;
        if (hasNext) wnext = *(const v16bf*)(wsrc + k0 + 32);

        // stream-prefetch x ~1KB ahead per row
        if (k0 + 256 < DK) {
            __builtin_prefetch(xrow0 + k0 + 256, 0, 3);
            __builtin_prefetch(xrow1 + k0 + 256, 0, 3);
        }

        // ---- load & convert two A fragments (16x32 bf16 each) ----
        const f4 p0 = *(const f4*)(xrow0 + k0 + akoff);
        const f4 p1 = *(const f4*)(xrow0 + k0 + akoff + 4);
        const f4 p2 = *(const f4*)(xrow0 + k0 + akoff + 8);
        const f4 p3 = *(const f4*)(xrow0 + k0 + akoff + 12);
        const f4 q0 = *(const f4*)(xrow1 + k0 + akoff);
        const f4 q1 = *(const f4*)(xrow1 + k0 + akoff + 4);
        const f4 q2 = *(const f4*)(xrow1 + k0 + akoff + 8);
        const f4 q3 = *(const f4*)(xrow1 + k0 + akoff + 12);

        v16bf a0, a1;
        a0[0]=f2bf(p0.x); a0[1]=f2bf(p0.y); a0[2]=f2bf(p0.z); a0[3]=f2bf(p0.w);
        a0[4]=f2bf(p1.x); a0[5]=f2bf(p1.y); a0[6]=f2bf(p1.z); a0[7]=f2bf(p1.w);
        a0[8]=f2bf(p2.x); a0[9]=f2bf(p2.y); a0[10]=f2bf(p2.z); a0[11]=f2bf(p2.w);
        a0[12]=f2bf(p3.x); a0[13]=f2bf(p3.y); a0[14]=f2bf(p3.z); a0[15]=f2bf(p3.w);
        a1[0]=f2bf(q0.x); a1[1]=f2bf(q0.y); a1[2]=f2bf(q0.z); a1[3]=f2bf(q0.w);
        a1[4]=f2bf(q1.x); a1[5]=f2bf(q1.y); a1[6]=f2bf(q1.z); a1[7]=f2bf(q1.w);
        a1[8]=f2bf(q2.x); a1[9]=f2bf(q2.y); a1[10]=f2bf(q2.z); a1[11]=f2bf(q2.w);
        a1[12]=f2bf(q3.x); a1[13]=f2bf(q3.y); a1[14]=f2bf(q3.z); a1[15]=f2bf(q3.w);

        // ---- 16 WMMAs: B fragments from LDS via two live buffers (bA, bB) ----
        const __bf16* sb = &smem[p][boff];
        v16bf bA = *(const v16bf*)(sb);                         // expert 0
        v16bf bB = *(const v16bf*)(sb + 1 * (RN * WROW));       // expert 1
#pragma unroll
        for (int e = 0; e < EN; e += 2) {
            acc0[e] = __builtin_amdgcn_wmma_f32_16x16x32_bf16(
                false, a0, false, bA, (short)0, acc0[e], false, false);
            acc1[e] = __builtin_amdgcn_wmma_f32_16x16x32_bf16(
                false, a1, false, bA, (short)0, acc1[e], false, false);
            if (e + 2 < EN)
                bA = *(const v16bf*)(sb + (e + 2) * (RN * WROW));
            acc0[e + 1] = __builtin_amdgcn_wmma_f32_16x16x32_bf16(
                false, a0, false, bB, (short)0, acc0[e + 1], false, false);
            acc1[e + 1] = __builtin_amdgcn_wmma_f32_16x16x32_bf16(
                false, a1, false, bB, (short)0, acc1[e + 1], false, false);
            if (e + 3 < EN)
                bB = *(const v16bf*)(sb + (e + 3) * (RN * WROW));
        }

        // land next W slice into the other LDS buffer (direct LDS addrspace store)
        if (hasNext) *(v16bf*)&smem[p ^ 1][stageOff] = wnext;
        __syncthreads();
    }

    // ---- store D: VGPR v, lane -> row v+8*half, col lmod ----
#pragma unroll
    for (int e = 0; e < EN; ++e) {
        float* op0 = Out + ((size_t)e * MTOT + m0) * RN + lmod;
        float* op1 = op0 + 16 * RN;
#pragma unroll
        for (int v = 0; v < 8; ++v) {
            const int row = v + half * 8;
            op0[(size_t)row * RN] = acc0[e][v];
            op1[(size_t)row * RN] = acc1[e][v];
        }
    }
}

// ---------------------------------------------------------------------------
extern "C" void kernel_launch(void* const* d_in, const int* in_sizes, int n_in,
                              void* d_out, int out_size, void* d_ws, size_t ws_size,
                              hipStream_t stream) {
    (void)in_sizes; (void)n_in; (void)out_size; (void)ws_size;

    const float* x   = (const float*)d_in[0];   // [B,S,DK] fp32
    const float* W   = (const float*)d_in[1];   // [E,RN,DK] fp32
    float*       out = (float*)d_out;           // [E, B*S, RN] fp32
    __bf16*      Wb  = (__bf16*)d_ws;           // 1 MB scratch

    // 1) W fp32 -> bf16 (524288 elems, 4/thread)
    w_cvt_bf16<<<dim3((EN * RN * DK / 4) / 256), dim3(256), 0, stream>>>(W, Wb);

    // 2) grouped GEMM: 512 waves x M=32 tokens, 8 waves/block -> 64 blocks
    lora_wmma_bf16<<<dim3(64), dim3(256), 0, stream>>>(x, Wb, out);
}
